// CrossDomainRAG_46832323395755
// MI455X (gfx1250) — compile-verified, matrix operand order checked by reading, not run on hardware
//
#include <hip/hip_runtime.h>
#include <hip/hip_bf16.h>

typedef __attribute__((ext_vector_type(16))) __bf16 v16bf;
typedef __attribute__((ext_vector_type(8)))  float  v8f;
typedef __attribute__((ext_vector_type(4)))  unsigned int v4u;
typedef __attribute__((ext_vector_type(4)))  int  v4i;
typedef __attribute__((ext_vector_type(8)))  int  v8i;

union V16B { v16bf v; uint4 u[2]; };

static __device__ inline v16bf ld_tile(const __bf16* p0, const __bf16* p1) {
    V16B t;
    t.u[0] = *(const uint4*)p0;
    t.u[1] = *(const uint4*)p1;
    return t.v;
}

// ---------------------------------------------------------------------------
// 1) q = hex_weights @ Wq_hex, then L2-normalize rows, emit bf16 (16384 x 64)
// ---------------------------------------------------------------------------
__global__ void k_hexproj(const float* __restrict__ hexw,
                          const float* __restrict__ Wqh,
                          __bf16* __restrict__ qn) {
    const int row = blockIdx.x;
    const int t   = threadIdx.x;          // 64 threads
    __shared__ float sh[64];
    __shared__ float red[64];
    sh[t] = hexw[row * 64 + t];
    __syncthreads();
    float acc = 0.f;
#pragma unroll 8
    for (int i = 0; i < 64; ++i) acc = fmaf(sh[i], Wqh[i * 64 + t], acc);
    red[t] = acc * acc;
    __syncthreads();
    for (int s = 32; s > 0; s >>= 1) {
        if (t < s) red[t] += red[t + s];
        __syncthreads();
    }
    const float rn = 1.f / fmaxf(sqrtf(red[0]), 1e-12f);
    qn[row * 64 + t] = (__bf16)(acc * rn);
}

// ---------------------------------------------------------------------------
// 2) L2-normalize doc_keys rows, emit bf16 (4096 x 64)
// ---------------------------------------------------------------------------
__global__ void k_docnorm(const float* __restrict__ dk,
                          __bf16* __restrict__ kn) {
    const int row = blockIdx.x;
    const int t   = threadIdx.x;          // 64 threads
    __shared__ float red[64];
    const float v = dk[row * 64 + t];
    red[t] = v * v;
    __syncthreads();
    for (int s = 32; s > 0; s >>= 1) {
        if (t < s) red[t] += red[t + s];
        __syncthreads();
    }
    const float rn = 1.f / fmaxf(sqrtf(red[0]), 1e-12f);
    kn[row * 64 + t] = (__bf16)(v * rn);
}

// ---------------------------------------------------------------------------
// 3) 1024x1024x1024 fp32 GEMM (weight fusion; tiny, run twice).
//    modeT==0: outF[r][c] = acc (fp32)
//    modeT==1: outBT[c][r] = bf16(acc)   (transposed store for WMMA B-tiles)
// ---------------------------------------------------------------------------
__global__ void k_gemm1k_f32(const float* __restrict__ A,
                             const float* __restrict__ B,
                             float* __restrict__ outF,
                             __bf16* __restrict__ outBT,
                             int modeT) {
    __shared__ float As[16][17];
    __shared__ float Bs[16][17];
    const int tx = threadIdx.x, ty = threadIdx.y;
    const int row = blockIdx.y * 16 + ty;
    const int col = blockIdx.x * 16 + tx;
    float acc = 0.f;
    for (int kt = 0; kt < 1024; kt += 16) {
        As[ty][tx] = A[row * 1024 + kt + tx];
        Bs[ty][tx] = B[(kt + ty) * 1024 + col];
        __syncthreads();
#pragma unroll
        for (int i = 0; i < 16; ++i) acc = fmaf(As[ty][i], Bs[i][tx], acc);
        __syncthreads();
    }
    if (modeT) outBT[(size_t)col * 1024 + row] = (__bf16)acc;
    else       outF[(size_t)row * 1024 + col] = acc;
}

// ---------------------------------------------------------------------------
// 4) bfused = bv @ Wo + bo   (1024 outputs)
// ---------------------------------------------------------------------------
__global__ void k_bfuse(const float* __restrict__ bv,
                        const float* __restrict__ Wo,
                        const float* __restrict__ bo,
                        float* __restrict__ bf) {
    const int n = blockIdx.x * 256 + threadIdx.x;
    if (n < 1024) {
        float acc = bo[n];
        for (int k = 0; k < 1024; ++k) acc = fmaf(bv[k], Wo[k * 1024 + n], acc);
        bf[n] = acc;
    }
}

// ---------------------------------------------------------------------------
// 5) Similarity (bf16 WMMA, K=64) + per-row top-8 + softmax.
//    One wave (32 threads) per 16-token tile; loops over 256 doc tiles.
// ---------------------------------------------------------------------------
__global__ void k_simtopk(const __bf16* __restrict__ qn,
                          const __bf16* __restrict__ kn,
                          int* __restrict__ topI,
                          float* __restrict__ topW) {
    const int lane  = threadIdx.x;        // 32 threads
    const int mBase = blockIdx.x * 16;
    const int mm = lane & 15;
    const int hf = lane >> 4;

    // A tile: 16 rows x K=64 of normalized q (two K=32 halves)
    const __bf16* qrow = qn + (size_t)(mBase + mm) * 64 + hf * 8;
    const v16bf a0 = ld_tile(qrow,      qrow + 16);
    const v16bf a1 = ld_tile(qrow + 32, qrow + 48);

    __shared__ float sh[256];

    float ts[8];
    int   ti[8];
#pragma unroll
    for (int u = 0; u < 8; ++u) { ts[u] = -3e38f; ti[u] = 0; }
    float tmin = -3e38f;

    for (int nt = 0; nt < 256; ++nt) {
        const int docBase = nt * 16;
        const __bf16* krow = kn + (size_t)(docBase + mm) * 64 + hf * 8;
        const v16bf b0 = ld_tile(krow,      krow + 16);
        const v16bf b1 = ld_tile(krow + 32, krow + 48);
        v8f c = {};
        c = __builtin_amdgcn_wmma_f32_16x16x32_bf16(false, a0, false, b0,
                                                    (short)0, c, false, false);
        c = __builtin_amdgcn_wmma_f32_16x16x32_bf16(false, a1, false, b1,
                                                    (short)0, c, false, false);
        // C layout: lane holds N=mm, rows M=r+8*hf in VGPR r
#pragma unroll
        for (int r = 0; r < 8; ++r) sh[(r + 8 * hf) * 16 + mm] = c[r];
        __syncthreads();
        if (lane < 16) {
#pragma unroll 4
            for (int j = 0; j < 16; ++j) {
                const float s = sh[lane * 16 + j];
                if (s > tmin) {                  // fast reject path
                    int am = 0; float mv = ts[0];
#pragma unroll
                    for (int u = 1; u < 8; ++u) if (ts[u] < mv) { mv = ts[u]; am = u; }
                    ts[am] = s; ti[am] = docBase + j;
                    mv = ts[0];
#pragma unroll
                    for (int u = 1; u < 8; ++u) mv = fminf(mv, ts[u]);
                    tmin = mv;
                }
            }
        }
        __syncthreads();
    }

    if (lane < 16) {
        float mx = ts[0];
#pragma unroll
        for (int u = 1; u < 8; ++u) mx = fmaxf(mx, ts[u]);
        float e[8], sum = 0.f;
#pragma unroll
        for (int u = 0; u < 8; ++u) { e[u] = __expf(ts[u] - mx); sum += e[u]; }
        const float inv = 1.f / sum;
        const int g = mBase + lane;
#pragma unroll
        for (int u = 0; u < 8; ++u) {
            topI[g * 8 + u] = ti[u];
            topW[g * 8 + u] = e[u] * inv;
        }
    }
}

// ---------------------------------------------------------------------------
// 6) retrieved = sum_k w_k * doc_values[idx_k]  ->  bf16 (16384 x 1024)
//    doc_values (16MB) is L2-resident, gather is cheap.
// ---------------------------------------------------------------------------
__global__ void k_gather(const float* __restrict__ docV,
                         const int* __restrict__ topI,
                         const float* __restrict__ topW,
                         __bf16* __restrict__ retBF) {
    const int row = blockIdx.x;
    const int t   = threadIdx.x;          // 256 threads, 4 dims each
    __shared__ int   shI[8];
    __shared__ float shW[8];
    if (t < 8) { shI[t] = topI[row * 8 + t]; shW[t] = topW[row * 8 + t]; }
    __syncthreads();
    const int d0 = t * 4;
    float4 acc = make_float4(0.f, 0.f, 0.f, 0.f);
#pragma unroll
    for (int k = 0; k < 8; ++k) {
        const float4 dv = *(const float4*)&docV[(size_t)shI[k] * 1024 + d0];
        const float w = shW[k];
        acc.x = fmaf(w, dv.x, acc.x);
        acc.y = fmaf(w, dv.y, acc.y);
        acc.z = fmaf(w, dv.z, acc.z);
        acc.w = fmaf(w, dv.w, acc.w);
    }
    union { __bf16 b[4]; uint2 u; } o;
    o.b[0] = (__bf16)acc.x; o.b[1] = (__bf16)acc.y;
    o.b[2] = (__bf16)acc.z; o.b[3] = (__bf16)acc.w;
    *(uint2*)&retBF[(size_t)row * 1024 + d0] = o.u;
}

// ---------------------------------------------------------------------------
// 7) out = x + sigmoid(gate) * (retrieved @ Wfused + bfused)
//    128x128 block, 8 waves (each owns a 16-row strip, shares the B panel).
//    A panel: global_load_async_to_lds_b128 (ASYNCcnt, per-lane).
//    B panel: TENSOR_LOAD_TO_LDS via TDM descriptor, wave 0 only (TENSORcnt).
//    Double-buffered; WMMA fragments fed by ds_load_b128.
// ---------------------------------------------------------------------------
#if __has_builtin(__builtin_amdgcn_tensor_load_to_lds)
#define USE_TDM 1
#else
#define USE_TDM 0
#endif

__global__ __launch_bounds__(256) void k_outgemm(
        const __bf16* __restrict__ retBF,
        const __bf16* __restrict__ WfT,     // transposed fused weights [n][k]
        const float*  __restrict__ bfused,
        const float*  __restrict__ x,
        const float*  __restrict__ gate,
        float* __restrict__ out) {
    const int tid  = threadIdx.x;
    const int lane = tid & 31;
    const int wave = tid >> 5;
    const int mBlock = blockIdx.x * 128;
    const int nBlock = blockIdx.y * 128;
    const int mm = lane & 15;
    const int hf = lane >> 4;

    __shared__ __align__(16) __bf16 Abuf[2][128 * 32];   // 8KB each
    __shared__ __align__(16) __bf16 Bbuf[2][128 * 32];   // 8KB each

    v8f acc[8];
#pragma unroll
    for (int t = 0; t < 8; ++t) acc[t] = (v8f){};

    // Stage one 128x32 A panel (async-to-LDS) + one 128x32 B panel (TDM).
    auto stage = [&](int kb, int buf) {
        // A: chunk c (0..511) covers row = c>>2, k-offset = (c&3)*8; 16B each.
#pragma unroll
        for (int j = 0; j < 2; ++j) {
            const int c   = tid + j * 256;
            const int row = c >> 2;
            const int ko  = (c & 3) * 8;
            const uint64_t ga = (uint64_t)(uintptr_t)
                (retBF + (size_t)(mBlock + row) * 1024 + kb + ko);
            const uint32_t la = (uint32_t)(uintptr_t)(&Abuf[buf][c * 8]);
            asm volatile("global_load_async_to_lds_b128 %0, %1, off"
                         :: "v"(la), "v"(ga) : "memory");
        }
#if USE_TDM
        if (wave == 0) {
            // D# for a 2-D 32(k) x 128(n) bf16 tile of WfT (1024x1024, 2B elems)
            const uint64_t gb = (uint64_t)(uintptr_t)
                (WfT + (size_t)nBlock * 1024 + kb);
            const uint32_t lb = (uint32_t)(uintptr_t)(&Bbuf[buf][0]);
            v4u g0;
            g0.x = 1u;                                    // count=1, user mode
            g0.y = lb;                                    // lds_addr
            g0.z = (unsigned int)gb;                      // global_addr[31:0]
            g0.w = (unsigned int)((gb >> 32) & 0x01FFFFFFu) | (2u << 30); // +type=2
            v8i g1 = {0, 0, 0, 0, 0, 0, 0, 0};
            g1[0] = (int)(1u << 16);                      // data_size=1 (2 bytes)
            g1[1] = (int)(1024u << 16);                   // tensor_dim0 = 1024 (k)
            g1[2] = (int)(1024u << 16);                   // tensor_dim1 = 1024 (n)
            g1[3] = (int)(32u << 16);                     // tile_dim0 = 32 (k)
            g1[4] = (int)128;                             // tile_dim1 = 128 (n)
            g1[5] = (int)1024;                            // tensor_dim0_stride
            const v4i z4 = {0, 0, 0, 0};
#if __clang_major__ >= 23
            const v8i z8 = {0, 0, 0, 0, 0, 0, 0, 0};
            __builtin_amdgcn_tensor_load_to_lds(g0, g1, z4, z4, z8, 0);
#else
            __builtin_amdgcn_tensor_load_to_lds(g0, g1, z4, z4, 0);
#endif
        }
#else
#pragma unroll
        for (int j = 0; j < 2; ++j) {
            const int c   = tid + j * 256;
            const int row = c >> 2;
            const int ko  = (c & 3) * 8;
            const uint64_t gb = (uint64_t)(uintptr_t)
                (WfT + (size_t)(nBlock + row) * 1024 + kb + ko);
            const uint32_t lb = (uint32_t)(uintptr_t)(&Bbuf[buf][c * 8]);
            asm volatile("global_load_async_to_lds_b128 %0, %1, off"
                         :: "v"(lb), "v"(gb) : "memory");
        }
#endif
    };

    stage(0, 0);
    for (int kt = 0; kt < 32; ++kt) {
        const int buf = kt & 1;
        asm volatile("s_wait_asynccnt 0x0" ::: "memory");  // this wave's A chunks
#if USE_TDM
        if (wave == 0) __builtin_amdgcn_s_wait_tensorcnt(0);   // B DMA done
#endif
        __syncthreads();                                   // panel visible to all
        if (kt + 1 < 32) stage((kt + 1) * 32, buf ^ 1);    // overlap with compute

        const __bf16* Ash = &Abuf[buf][(wave * 16 + mm) * 32 + hf * 8];
        const v16bf a = ld_tile(Ash, Ash + 16);
#pragma unroll
        for (int t = 0; t < 8; ++t) {
            const __bf16* Bsh = &Bbuf[buf][(t * 16 + mm) * 32 + hf * 8];
            const v16bf b = ld_tile(Bsh, Bsh + 16);
            acc[t] = __builtin_amdgcn_wmma_f32_16x16x32_bf16(
                false, a, false, b, (short)0, acc[t], false, false);
        }
        __syncthreads();   // reads of `buf` done before it is restaged next iter
    }

    const float sig = 1.f / (1.f + __expf(-gate[0]));
#pragma unroll
    for (int t = 0; t < 8; ++t) {
        const int col = nBlock + t * 16 + mm;
        const float bcol = bfused[col];
#pragma unroll
        for (int r = 0; r < 8; ++r) {
            const int row = mBlock + wave * 16 + r + 8 * hf;
            const size_t o = (size_t)row * 1024 + col;
            out[o] = x[o] + sig * (acc[t][r] + bcol);
        }
    }
}

// ---------------------------------------------------------------------------
extern "C" void kernel_launch(void* const* d_in, const int* in_sizes, int n_in,
                              void* d_out, int out_size, void* d_ws, size_t ws_size,
                              hipStream_t stream) {
    (void)in_sizes; (void)n_in; (void)out_size; (void)ws_size;
    const float* x     = (const float*)d_in[0];
    const float* hexw  = (const float*)d_in[1];
    const float* dkeys = (const float*)d_in[2];
    const float* dvals = (const float*)d_in[3];
    const float* Wqh   = (const float*)d_in[4];
    // d_in[5]=Wq, d_in[6]=Wk, d_in[8]=bq, d_in[9]=bk: provably unused (1-key attn)
    const float* Wv    = (const float*)d_in[7];
    const float* bv    = (const float*)d_in[10];
    const float* Wo    = (const float*)d_in[11];
    const float* bo    = (const float*)d_in[12];
    const float* Wdoc  = (const float*)d_in[13];
    const float* gate  = (const float*)d_in[14];
    float* out = (float*)d_out;

    // workspace carve-out (256B aligned slices)
    char* ws = (char*)d_ws;
    auto carve = [&](size_t bytes) -> char* {
        char* p = ws;
        ws += (bytes + 255) & ~(size_t)255;
        return p;
    };
    __bf16* qn    = (__bf16*)carve((size_t)16384 * 64 * 2);
    __bf16* kn    = (__bf16*)carve((size_t)4096 * 64 * 2);
    int*    topI  = (int*)   carve((size_t)16384 * 8 * 4);
    float*  topW  = (float*) carve((size_t)16384 * 8 * 4);
    __bf16* retBF = (__bf16*)carve((size_t)16384 * 1024 * 2);
    float*  T1    = (float*) carve((size_t)1024 * 1024 * 4);
    __bf16* WfT   = (__bf16*)carve((size_t)1024 * 1024 * 2);
    float*  bfsd  = (float*) carve((size_t)1024 * 4);

    k_hexproj<<<16384, 64, 0, stream>>>(hexw, Wqh, qn);
    k_docnorm<<<4096, 64, 0, stream>>>(dkeys, kn);
    k_gemm1k_f32<<<dim3(64, 64), dim3(16, 16), 0, stream>>>(Wdoc, Wv, T1, nullptr, 0);
    k_gemm1k_f32<<<dim3(64, 64), dim3(16, 16), 0, stream>>>(T1, Wo, nullptr, WfT, 1);
    k_bfuse<<<4, 256, 0, stream>>>(bv, Wo, bo, bfsd);
    k_simtopk<<<1024, 32, 0, stream>>>(qn, kn, topI, topW);
    k_gather<<<16384, 256, 0, stream>>>(dvals, topI, topW, retBF);
    k_outgemm<<<dim3(128, 8), 256, 0, stream>>>(retBF, WfT, bfsd, x, gate, out);
}
